// ArcFaceLoss_35029753266820
// MI455X (gfx1250) — matrix operand order repaired and checked by reading, not compiled, not run on hardware
//
#include <hip/hip_runtime.h>
#include <math.h>
#include <stdint.h>

// ---------- types ----------
typedef __bf16 bf16_t;
typedef __attribute__((ext_vector_type(16))) __bf16 v16bf;
typedef __attribute__((ext_vector_type(8)))  __bf16 v8bf;
typedef __attribute__((ext_vector_type(8)))  float  v8f;

// ---------- ArcFace constants ----------
#define S_SCALE 64.0f
#define COS_M   0.8775825618903728f
#define SIN_M   0.479425538604203f
#define TH_C   (-0.8775825618903728f)
#define MM_C    0.2397127693021015f
#define EPS_C   1e-7f

// ---------- problem sizes ----------
constexpr int Bv = 1024;     // batch
constexpr int Dv = 512;      // feature dim
constexpr int Cv = 100000;   // classes
constexpr int CT = 32;                  // class columns per block
constexpr int NBLK = Cv / CT;           // 3125 blocks
constexpr int ROWS_PER_PASS = 256;      // 8 waves x 32 rows
constexpr int NPASS = Bv / ROWS_PER_PASS; // 4

// ---------- fragment load helpers (per ISA 7.12.2 layouts) ----------
__device__ __forceinline__ v16bf load_frag_a(const bf16_t* p) {
  // per lane: 8 consecutive bf16 at K=8h, and 8 more at K=16+8h
  v8bf a0 = *(const v8bf*)(p);
  v8bf a1 = *(const v8bf*)(p + 16);
  return __builtin_shufflevector(a0, a1, 0,1,2,3,4,5,6,7,8,9,10,11,12,13,14,15);
}
__device__ __forceinline__ v16bf load_frag_b(const bf16_t* p) {
  // per lane: 8 consecutive bf16 at K=8h, and 8 more at K=16+8h
  v8bf b0 = *(const v8bf*)(p);
  v8bf b1 = *(const v8bf*)(p + 16);
  return __builtin_shufflevector(b0, b1, 0,1,2,3,4,5,6,7,8,9,10,11,12,13,14,15);
}
#define WMMA(A, B, C) \
  __builtin_amdgcn_wmma_f32_16x16x32_bf16(false, (A), false, (B), (short)0, (C), false, false)

// =====================================================================
// Kernel 1: normalize embeddings, split to bf16 hi/lo; zero the output.
// =====================================================================
__global__ void prep_e_kernel(const float* __restrict__ e,
                              bf16_t* __restrict__ ehi,
                              bf16_t* __restrict__ elo,
                              float* __restrict__ out) {
  if (blockIdx.x == 0 && threadIdx.x == 0) out[0] = 0.0f;
  const int wave = threadIdx.x >> 5, lane = threadIdx.x & 31;
  const int row = blockIdx.x * 8 + wave;
  if (row >= Bv) return;
  const float* er = e + (size_t)row * Dv;
  float ss = 0.f;
  for (int k = lane; k < Dv; k += 32) { float v = er[k]; ss += v * v; }
  #pragma unroll
  for (int m = 16; m >= 1; m >>= 1) ss += __shfl_xor(ss, m);
  const float inv = 1.0f / fmaxf(sqrtf(ss), 1e-12f);
  for (int k = lane; k < Dv; k += 32) {
    float v = er[k] * inv;
    bf16_t h = (bf16_t)v;
    ehi[(size_t)row * Dv + k] = h;
    elo[(size_t)row * Dv + k] = (bf16_t)(v - (float)h);
  }
}

// =====================================================================
// Kernel 2: exact fp32 target logits: tgt[b] = {S*cos, S*phi}.
// =====================================================================
__global__ void target_kernel(const float* __restrict__ e,
                              const int* __restrict__ labels,
                              const float* __restrict__ w,
                              float* __restrict__ tgt) {
  const int wave = threadIdx.x >> 5, lane = threadIdx.x & 31;
  const int row = blockIdx.x * 8 + wave;
  if (row >= Bv) return;
  const int c = labels[row];
  const float* er = e + (size_t)row * Dv;
  const float* wr = w + (size_t)c * Dv;
  float dot = 0.f, se = 0.f, sw = 0.f;
  for (int k = lane; k < Dv; k += 32) {
    float a = er[k], b = wr[k];
    dot += a * b; se += a * a; sw += b * b;
  }
  #pragma unroll
  for (int m = 16; m >= 1; m >>= 1) {
    dot += __shfl_xor(dot, m);
    se  += __shfl_xor(se, m);
    sw  += __shfl_xor(sw, m);
  }
  if (lane == 0) {
    float inv  = 1.0f / (fmaxf(sqrtf(se), 1e-12f) * fmaxf(sqrtf(sw), 1e-12f));
    float cosv = fminf(fmaxf(dot * inv, -1.0f + EPS_C), 1.0f - EPS_C);
    float sinv = sqrtf(1.0f - cosv * cosv);
    float phi  = cosv * COS_M - sinv * SIN_M;
    phi = (cosv > TH_C) ? phi : (cosv - MM_C);
    tgt[row * 2 + 0] = S_SCALE * cosv;
    tgt[row * 2 + 1] = S_SCALE * phi;
  }
}

// =====================================================================
// Kernel 3: fused bf16-split WMMA GEMM + online row LSE partials.
// Block: 32 classes. Async-stage raw w tile to LDS (ASYNCcnt), build
// normalized bf16 hi/lo tiles, then each wave computes a 32x32 tile
// per pass (4 accumulators, 12 WMMA per k-step).
// k-loop unroll limited to 2 so the backend cannot hoist all 16 k-steps
// of B fragments into registers (round 2 spilled to scratch).
// =====================================================================
__global__ __launch_bounds__(256) void gemm_lse_kernel(
    const float* __restrict__ w,
    const bf16_t* __restrict__ ehi,
    const bf16_t* __restrict__ elo,
    float2* __restrict__ partial) {
  __shared__ float  lds_wraw[CT][Dv];  // 64 KB raw fp32 tile
  __shared__ bf16_t lds_whi[CT][Dv];   // 32 KB
  __shared__ bf16_t lds_wlo[CT][Dv];   // 32 KB
  __shared__ float  lds_invw[CT];

  const int tid  = threadIdx.x;
  const int wave = tid >> 5, lane = tid & 31;
  const int c_base = blockIdx.x * CT;

  // ---- Phase 0: async copy raw w tile (64 KB) global -> LDS ----
  {
    const char* gbase = (const char*)(w + (size_t)c_base * Dv);
    const char* lbase = (const char*)&lds_wraw[0][0];
    #pragma unroll
    for (int i = 0; i < (CT * Dv * 4) / (256 * 16); ++i) {  // 16 iters
      const int boff = (tid + i * 256) * 16;
      uint32_t loff = (uint32_t)(uintptr_t)(lbase + boff);  // flat->LDS offset
      const char* g = gbase + boff;
      asm volatile("global_load_async_to_lds_b128 %0, %1, off"
                   :: "v"(loff), "v"(g) : "memory");
    }
    asm volatile("s_wait_asynccnt 0x0" ::: "memory");
  }
  __syncthreads();

  // ---- Phase 1: per-class inverse norms from LDS (each wave: 4 rows) ----
  #pragma unroll
  for (int r = 0; r < 4; ++r) {
    const int cr = wave * 4 + r;
    float ss = 0.f;
    for (int k = lane; k < Dv; k += 32) { float v = lds_wraw[cr][k]; ss += v * v; }
    #pragma unroll
    for (int m = 16; m >= 1; m >>= 1) ss += __shfl_xor(ss, m);
    if (lane == 0) lds_invw[cr] = 1.0f / fmaxf(sqrtf(ss), 1e-12f);
  }
  __syncthreads();

  // ---- Phase 2: normalized bf16 hi/lo tiles from LDS raw tile ----
  #pragma unroll
  for (int i = 0; i < (CT * Dv) / 256; ++i) {
    const int idx = tid + i * 256;
    const int rr = idx >> 9, kk = idx & (Dv - 1);
    float v = lds_wraw[rr][kk] * lds_invw[rr];
    bf16_t h = (bf16_t)v;
    lds_whi[rr][kk] = h;
    lds_wlo[rr][kk] = (bf16_t)(v - (float)h);
  }
  __syncthreads();

  // ---- Phase 3: per wave 32(rows) x 32(cols) tile, 4 passes ----
  const int half = lane >> 4, lan = lane & 15;

  #pragma unroll 1
  for (int pass = 0; pass < NPASS; ++pass) {
    const int m_base = pass * ROWS_PER_PASS + wave * 32;
    v8f acc00 = {0,0,0,0,0,0,0,0}, acc01 = {0,0,0,0,0,0,0,0};
    v8f acc10 = {0,0,0,0,0,0,0,0}, acc11 = {0,0,0,0,0,0,0,0};

    const bf16_t* pa_hi0 = ehi + (size_t)(m_base + lan) * Dv + 8 * half;
    const bf16_t* pa_lo0 = elo + (size_t)(m_base + lan) * Dv + 8 * half;
    const bf16_t* pa_hi1 = pa_hi0 + (size_t)16 * Dv;
    const bf16_t* pa_lo1 = pa_lo0 + (size_t)16 * Dv;
    const bf16_t* pb_hi0 = &lds_whi[lan][8 * half];
    const bf16_t* pb_lo0 = &lds_wlo[lan][8 * half];
    const bf16_t* pb_hi1 = &lds_whi[16 + lan][8 * half];
    const bf16_t* pb_lo1 = &lds_wlo[16 + lan][8 * half];

    #pragma unroll 2
    for (int ks = 0; ks < Dv / 32; ++ks) {
      const int kk = ks * 32;
      v16bf ahi0 = load_frag_a(pa_hi0 + kk), alo0 = load_frag_a(pa_lo0 + kk);
      v16bf ahi1 = load_frag_a(pa_hi1 + kk), alo1 = load_frag_a(pa_lo1 + kk);
      v16bf bhi0 = load_frag_b(pb_hi0 + kk), blo0 = load_frag_b(pb_lo0 + kk);
      v16bf bhi1 = load_frag_b(pb_hi1 + kk), blo1 = load_frag_b(pb_lo1 + kk);

      acc00 = WMMA(ahi0, bhi0, acc00);
      acc01 = WMMA(ahi0, bhi1, acc01);
      acc10 = WMMA(ahi1, bhi0, acc10);
      acc11 = WMMA(ahi1, bhi1, acc11);
      acc00 = WMMA(ahi0, blo0, acc00);
      acc01 = WMMA(ahi0, blo1, acc01);
      acc10 = WMMA(ahi1, blo0, acc10);
      acc11 = WMMA(ahi1, blo1, acc11);
      acc00 = WMMA(alo0, bhi0, acc00);
      acc01 = WMMA(alo0, bhi1, acc01);
      acc10 = WMMA(alo1, bhi0, acc10);
      acc11 = WMMA(alo1, bhi1, acc11);
    }

    // Per-row LSE over this wave's full 32 columns (no cross-wave merge).
    #pragma unroll
    for (int rt = 0; rt < 2; ++rt) {
      #pragma unroll
      for (int r = 0; r < 8; ++r) {
        float c0 = (rt == 0) ? acc00[r] : acc10[r];
        float c1 = (rt == 0) ? acc01[r] : acc11[r];
        float v0 = S_SCALE * fminf(fmaxf(c0, -1.0f + EPS_C), 1.0f - EPS_C);
        float v1 = S_SCALE * fminf(fmaxf(c1, -1.0f + EPS_C), 1.0f - EPS_C);
        float rmax = fmaxf(v0, v1);
        rmax = fmaxf(rmax, __shfl_xor(rmax, 1));
        rmax = fmaxf(rmax, __shfl_xor(rmax, 2));
        rmax = fmaxf(rmax, __shfl_xor(rmax, 4));
        rmax = fmaxf(rmax, __shfl_xor(rmax, 8));
        float ev = __expf(v0 - rmax) + __expf(v1 - rmax);
        ev += __shfl_xor(ev, 1);
        ev += __shfl_xor(ev, 2);
        ev += __shfl_xor(ev, 4);
        ev += __shfl_xor(ev, 8);
        if (lan == 0) {
          const int grow = m_base + rt * 16 + 8 * half + r;
          partial[(size_t)blockIdx.x * Bv + grow] = make_float2(rmax, ev);
        }
      }
    }
  }
}

// =====================================================================
// Kernel 4: merge partials per row, apply target correction, mean NLL.
// =====================================================================
__global__ void reduce_kernel(const float2* __restrict__ partial,
                              const float* __restrict__ tgt,
                              float* __restrict__ out) {
  const int wave = threadIdx.x >> 5, lane = threadIdx.x & 31;
  const int row = blockIdx.x * 8 + wave;
  if (row >= Bv) return;
  float m = -3.0e38f, s = 0.f;
  for (int i = lane; i < NBLK; i += 32) {
    float2 p = partial[(size_t)i * Bv + row];
    float mm = fmaxf(m, p.x);
    s = s * __expf(m - mm) + p.y * __expf(p.x - mm);
    m = mm;
  }
  #pragma unroll
  for (int msk = 16; msk >= 1; msk >>= 1) {
    float om = __shfl_xor(m, msk);
    float os = __shfl_xor(s, msk);
    float mm = fmaxf(m, om);
    s = s * __expf(m - mm) + os * __expf(om - mm);
    m = mm;
  }
  if (lane == 0) {
    const float a = tgt[row * 2 + 0];   // S*cos at label
    const float p = tgt[row * 2 + 1];   // S*phi at label
    float s2 = s - __expf(a - m) + __expf(p - m);  // swap label term
    float nll = (m + logf(s2)) - p;
    atomicAdd(out, nll * (1.0f / (float)Bv));
  }
}

// =====================================================================
extern "C" void kernel_launch(void* const* d_in, const int* in_sizes, int n_in,
                              void* d_out, int out_size, void* d_ws, size_t ws_size,
                              hipStream_t stream) {
  const float* emb    = (const float*)d_in[0];   // [1024,512] f32
  const int*   labels = (const int*)d_in[1];     // [1024]
  const float* weight = (const float*)d_in[2];   // [100000,512] f32
  float* out = (float*)d_out;

  char* ws = (char*)d_ws;
  bf16_t* ehi = (bf16_t*)ws;                               // 1 MB
  bf16_t* elo = (bf16_t*)(ws + (size_t)Bv * Dv * 2);       // 1 MB
  float*  tgt = (float*)(ws + (size_t)Bv * Dv * 4);        // 8 KB
  float2* partial = (float2*)(ws + (size_t)Bv * Dv * 4 + (size_t)Bv * 2 * sizeof(float)); // 25.6 MB

  hipLaunchKernelGGL(prep_e_kernel, dim3(Bv / 8), dim3(256), 0, stream,
                     emb, ehi, elo, out);
  hipLaunchKernelGGL(target_kernel, dim3(Bv / 8), dim3(256), 0, stream,
                     emb, labels, weight, tgt);
  hipLaunchKernelGGL(gemm_lse_kernel, dim3(NBLK), dim3(256), 0, stream,
                     weight, ehi, elo, partial);
  hipLaunchKernelGGL(reduce_kernel, dim3(Bv / 8), dim3(256), 0, stream,
                     partial, tgt, out);
}
